// Sinkhorn_43069932044491
// MI455X (gfx1250) — matrix operand order repaired and checked by reading, not compiled, not run on hardware
//
#include <hip/hip_runtime.h>
#include <hip/hip_bf16.h>

typedef __attribute__((ext_vector_type(16))) _Float16 v16h;
typedef __attribute__((ext_vector_type(8)))  _Float16 v8h;
typedef __attribute__((ext_vector_type(8)))  float    v8f;

#define N_DIM 4096
#define S_DIM 4
#define XROWS 16                       // padded batch rows for B operand (cols 4..15 zero)
#define SCALE_U 2097152.0f             // 2^21 : keeps scaled potentials f16-normal
#define LOG_SCALE_U 14.556090791758851f // 21 * ln(2)

// ---------------- Kernel 1: K = exp(-C/eps) in f16, plus its transpose ----------------
__global__ void __launch_bounds__(256)
sinkhorn_build_K(const float* __restrict__ C, const float* __restrict__ epsp,
                 _Float16* __restrict__ K16, _Float16* __restrict__ KT16) {
    __shared__ _Float16 tile[32][33];
    const float inv_eps = 1.0f / epsp[0];
    const int x  = threadIdx.x & 31;   // column within tile
    const int yy = threadIdx.x >> 5;   // 0..7
    const int i0 = blockIdx.y * 32;
    const int j0 = blockIdx.x * 32;
#pragma unroll
    for (int ry = 0; ry < 4; ++ry) {
        int y = yy + ry * 8;
        float c = C[(size_t)(i0 + y) * N_DIM + (j0 + x)];
        _Float16 h = (_Float16)__expf(-c * inv_eps);
        K16[(size_t)(i0 + y) * N_DIM + (j0 + x)] = h;  // row-major K
        tile[x][y] = h;                                 // stage for transpose
    }
    __syncthreads();
#pragma unroll
    for (int ry = 0; ry < 4; ++ry) {
        int jj = yy + ry * 8;
        KT16[(size_t)(j0 + jj) * N_DIM + (i0 + x)] = tile[jj][x]; // coalesced KT
    }
}

// ---------------- Kernel 2: initialize scaled potentials (v0 = exp(g0/eps) = 1) --------
__global__ void sinkhorn_init_vec(_Float16* __restrict__ Ut, _Float16* __restrict__ Vt) {
    int idx = blockIdx.x * blockDim.x + threadIdx.x;  // exactly XROWS*N_DIM threads
    Ut[idx] = (_Float16)0.0f;                          // pad rows stay 0 forever
    Vt[idx] = (idx < S_DIM * N_DIM) ? (_Float16)1.0f : (_Float16)0.0f;
}

// ---------------- Kernel 3: one Sinkhorn half-step via WMMA ---------------------------
// y(16 rows x 4 cols) = M(16 x 4096) * xT^T ; out = w * 2^21 / y
// M: (4096 x 4096) f16 row-major (K or K^T). xT: (XROWS x 4096) f16, rows 0..3 live.
// w/outF: (4 x 4096) f32.  outT: (XROWS x 4096) f16, rows 0..3 written.
__global__ void __launch_bounds__(128)
sinkhorn_half_step(const _Float16* __restrict__ M,
                   const _Float16* __restrict__ xT,
                   const float*    __restrict__ w,
                   float*          __restrict__ outF,
                   _Float16*       __restrict__ outT) {
    const int lane = threadIdx.x & 31;
    const int wave = threadIdx.x >> 5;            // 0..3 (4 waves/block)
    const int i0   = blockIdx.x * 64 + wave * 16; // 16 rows per wave
    const int half = lane >> 4;                   // lane group 0/1
    const int n    = lane & 15;                   // output column (s for n<4)

    // A operand: lane covers row (i0+n); K-chunks half*8 .. half*8+7 and +16 (ISA 7.12.2)
    const _Float16* pa = M  + (size_t)(i0 + n) * N_DIM + half * 8;
    // B operand: lane covers column n; K-chunks half*8 .. half*8+7 and +16 (same as A)
    const _Float16* pb = xT + (size_t)n * N_DIM + half * 8;

    v8f acc = {};
#pragma unroll 4
    for (int j0 = 0; j0 < N_DIM; j0 += 32) {
        __builtin_prefetch((const void*)(pa + j0 + 512), 0, 0); // global_prefetch_b8
        v8h alo = *(const v8h*)(pa + j0);
        v8h ahi = *(const v8h*)(pa + j0 + 16);
        v8h blo = *(const v8h*)(pb + j0);
        v8h bhi = *(const v8h*)(pb + j0 + 16);
        v16h a = __builtin_shufflevector(alo, ahi, 0,1,2,3,4,5,6,7,8,9,10,11,12,13,14,15);
        v16h b = __builtin_shufflevector(blo, bhi, 0,1,2,3,4,5,6,7,8,9,10,11,12,13,14,15);
        acc = __builtin_amdgcn_wmma_f32_16x16x32_f16(false, a, false, b,
                                                     (short)0, acc, false, false);
    }
    // D layout: VGPR r holds row (r + 8*half), column = n  (ISA 7.12.2)
    if (n < S_DIM) {
#pragma unroll
        for (int r = 0; r < 8; ++r) {
            int row  = i0 + half * 8 + r;
            float o  = w[n * N_DIM + row] * SCALE_U / acc[r];
            outF[n * N_DIM + row]          = o;
            outT[(size_t)n * N_DIM + row]  = (_Float16)o;
        }
    }
}

// ---------------- Kernel 4: f = eps*(ln U - 21 ln2), g = eps*ln V ---------------------
__global__ void sinkhorn_finalize(const float* __restrict__ Uf, const float* __restrict__ Vf,
                                  const float* __restrict__ epsp, float* __restrict__ out) {
    int idx = blockIdx.x * blockDim.x + threadIdx.x;
    float e = epsp[0];
    if (idx < S_DIM * N_DIM)
        out[idx] = e * (__logf(Uf[idx]) - LOG_SCALE_U);
    else if (idx < 2 * S_DIM * N_DIM)
        out[idx] = e * __logf(Vf[idx - S_DIM * N_DIM]);
}

extern "C" void kernel_launch(void* const* d_in, const int* in_sizes, int n_in,
                              void* d_out, int out_size, void* d_ws, size_t ws_size,
                              hipStream_t stream) {
    const float* alpha = (const float*)d_in[0];  // (4, 4096)
    const float* beta  = (const float*)d_in[1];  // (4, 4096)
    const float* C     = (const float*)d_in[2];  // (4096, 4096)
    const float* eps   = (const float*)d_in[3];  // scalar

    char* ws = (char*)d_ws;
    _Float16* K16  = (_Float16*)ws;                                   // 32 MB
    _Float16* KT16 = (_Float16*)(ws + (size_t)32 * 1024 * 1024);      // 32 MB
    _Float16* Ut16 = (_Float16*)(ws + (size_t)64 * 1024 * 1024);      // 128 KB
    _Float16* Vt16 = Ut16 + XROWS * N_DIM;                            // 128 KB
    float*    Uf   = (float*)(Vt16 + XROWS * N_DIM);                  // 64 KB
    float*    Vf   = Uf + S_DIM * N_DIM;                              // 64 KB

    sinkhorn_build_K<<<dim3(N_DIM / 32, N_DIM / 32), 256, 0, stream>>>(C, eps, K16, KT16);
    sinkhorn_init_vec<<<(XROWS * N_DIM) / 256, 256, 0, stream>>>(Ut16, Vt16);

    for (int it = 0; it < 10; ++it) {
        // u = alpha / (K v)      (scaled by 2^21)
        sinkhorn_half_step<<<N_DIM / 64, 128, 0, stream>>>(K16,  Vt16, alpha, Uf, Ut16);
        // v = beta / (K^T u)     (scale cancels in the division)
        sinkhorn_half_step<<<N_DIM / 64, 128, 0, stream>>>(KT16, Ut16, beta,  Vf, Vt16);
    }

    sinkhorn_finalize<<<(2 * S_DIM * N_DIM) / 256, 256, 0, stream>>>(Uf, Vf, eps, (float*)d_out);
}
